// QuaternionLSTMCell_55018531062039
// MI455X (gfx1250) — matrix operand order repaired and checked by reading, not compiled, not run on hardware
//
#include <hip/hip_runtime.h>
#include <hip/hip_fp16.h>
#include <cstdint>

typedef __attribute__((ext_vector_type(16))) _Float16 v16h;
typedef __attribute__((ext_vector_type(8)))  float    v8f;
typedef int v4i __attribute__((vector_size(16)));

#define BQ   2048
#define IQ   512
#define HQ   512
#define KTOT 4096          // 4*I (x part) + 4*H (h part)
#define NTOT 8192          // 512 outputs * 16 (gate,comp)
#define SAS  40            // LDS row stride in halves (80B: 16B aligned, bank-spread)
#define KSTEPS (KTOT / 32) // 128

#if __has_builtin(__builtin_amdgcn_global_load_async_to_lds_b128)
#define QL_ASYNC 1
#else
#define QL_ASYNC 0
#endif

__device__ __forceinline__ void ql_wait_async0() {
#if __has_builtin(__builtin_amdgcn_s_wait_asynccnt)
  __builtin_amdgcn_s_wait_asynccnt(0);
#else
  asm volatile("s_wait_asynccnt 0x0" ::: "memory");
#endif
}

#if QL_ASYNC
typedef __attribute__((address_space(1))) v4i* ql_gptr;
typedef __attribute__((address_space(3))) v4i* ql_lptr;
__device__ __forceinline__ void ql_async_b128(const _Float16* g, _Float16* l) {
  __builtin_amdgcn_global_load_async_to_lds_b128(
      (ql_gptr)(g), (ql_lptr)(l), 0, 0);
}
#endif

// ---------------------------------------------------------------------------
// Pack A = [xf | hf] as f16, row-major [B=2048][K=4096]
// xf[b, cc*512 + i] = x[b, i, cc]; hf in columns 2048..4095.
// ---------------------------------------------------------------------------
__global__ __launch_bounds__(256) void qlstm_pack_a(
    const float* __restrict__ x, const float* __restrict__ h,
    _Float16* __restrict__ Ah)
{
  int idx = blockIdx.x * 256 + threadIdx.x;     // exactly 2048*4096 threads
  int b = idx >> 12;
  int k = idx & 4095;
  float v;
  if (k < 2048) {
    int cc = k >> 9, i = k & 511;
    v = x[((b << 9) + i) * 4 + cc];
  } else {
    int k2 = k - 2048;
    int cc = k2 >> 9, j = k2 & 511;
    v = h[((b << 9) + j) * 4 + cc];
  }
  Ah[idx] = (_Float16)v;
}

// ---------------------------------------------------------------------------
// Pack Hamilton weight matrix, f16, layout Bp[n][k] (K contiguous).
// n = o*16 + g*4 + cc  (g: 0=i,1=f,2=o,3=c ; cc = output quaternion component)
// M[rc*512+i, cc*512+o] = sign[rc][cc] * W[rc^cc][o][i]
// ---------------------------------------------------------------------------
__global__ __launch_bounds__(256) void qlstm_pack_b(
    const float* __restrict__ Wi, const float* __restrict__ Wf,
    const float* __restrict__ Wo, const float* __restrict__ Wc,
    const float* __restrict__ Ui, const float* __restrict__ Uf,
    const float* __restrict__ Uo, const float* __restrict__ Uc,
    _Float16* __restrict__ Bp)
{
  int idx = blockIdx.x * 256 + threadIdx.x;     // exactly 8192*4096 threads
  int n = idx >> 12;
  int k = idx & 4095;
  int o  = n >> 4;
  int g  = (n >> 2) & 3;
  int cc = n & 3;
  int k2 = k & 2047;
  int rc = k2 >> 9;
  int i  = k2 & 511;
  const float* W = (g == 0) ? Wi : (g == 1) ? Wf : (g == 2) ? Wo : Wc;
  const float* U = (g == 0) ? Ui : (g == 1) ? Uf : (g == 2) ? Uo : Uc;
  const float* src = (k < 2048) ? W : U;
  int   comp = rc ^ cc;
  float sgn  = ((0x5390 >> ((rc << 2) + cc)) & 1) ? -1.0f : 1.0f;
  float v = sgn * src[comp * (HQ * IQ) + o * IQ + i];
  Bp[idx] = (_Float16)v;
}

// ---------------------------------------------------------------------------
// Fused GEMM (f16 WMMA, f32 accum) + LSTM gate epilogue.
// Block tile: 128 (batch) x 256 (columns = 16 outputs * 16 gate/comp).
// 8 waves = wm(2) x wn(4); each wave: 64x64 tile = 4x4 WMMA tiles.
// Double-buffered LDS; global->LDS staging via async copies; loop written
// as 64 iterations x 2 static phases with unroll pinned to 1 so the
// backend cannot over-pipeline and overflow the 256-VGPR file.
// ---------------------------------------------------------------------------
__device__ __forceinline__ float sigm(float v) { return 1.0f / (1.0f + expf(-v)); }

__global__ __launch_bounds__(256) void qlstm_gemm(
    const _Float16* __restrict__ Ah, const _Float16* __restrict__ Bp,
    const float* __restrict__ c_in,
    const float* __restrict__ bi, const float* __restrict__ bf,
    const float* __restrict__ bo, const float* __restrict__ bc,
    float* __restrict__ out_h, float* __restrict__ out_c)
{
  __shared__ _Float16 sA[2][128 * SAS];   // 2 x 10240 B
  __shared__ _Float16 sB[2][256 * SAS];   // 2 x 20480 B

  const int tid  = threadIdx.x;
  const int nb   = blockIdx.x;   // 0..31  (column blocks: 16 outputs each)
  const int mb   = blockIdx.y;   // 0..15  (batch blocks)
  const int wave = tid >> 5;
  const int lane = tid & 31;
  const int wm   = wave >> 2;    // 0..1   (64-row slab)
  const int wn   = wave & 3;     // 0..3   (64-col slab)
  const int lrow  = lane & 15;
  const int lhalf = lane >> 4;   // 0/1

  // Per-thread staging coordinates: 32-bit element offsets off uniform bases.
  const int rowA0 = tid >> 2;            // A chunk 0: rows 0..63
  const int rowA1 = rowA0 + 64;          // A chunk 1: rows 64..127
  const int offA  = (tid & 3) << 3;      // 0,8,16,24 halves
  const int aOff0 = (mb * 128 + rowA0) * KTOT + offA;
  const int aOff1 = (mb * 128 + rowA1) * KTOT + offA;
  const int lA0   = rowA0 * SAS + offA;
  const int lA1   = rowA1 * SAS + offA;
  int bOff[4], lB[4];
#pragma unroll
  for (int c = 0; c < 4; ++c) {
    int rowB = (tid >> 2) + c * 64;      // 0..255
    bOff[c] = (nb * 256 + rowB) * KTOT + offA;
    lB[c]   = rowB * SAS + offA;
  }

  v8f acc[4][4];
#pragma unroll
  for (int mt = 0; mt < 4; ++mt)
#pragma unroll
    for (int nt = 0; nt < 4; ++nt) acc[mt][nt] = 0.0f;

  union Frag { v16h h; uint4 q[2]; };

  // One phase = one 32-deep K-step against static buffer `P`.
#define QL_FETCH(step, P)                                              \
  {                                                                    \
    int ko = (step) * 32;                                              \
    ql_async_b128(Ah + aOff0 + ko, sA[P] + lA0);                       \
    ql_async_b128(Ah + aOff1 + ko, sA[P] + lA1);                       \
    _Pragma("unroll")                                                  \
    for (int c = 0; c < 4; ++c)                                        \
      ql_async_b128(Bp + bOff[c] + ko, sB[P] + lB[c]);                 \
  }

#define QL_FETCH_SYNC(step, P)                                         \
  {                                                                    \
    int ko = (step) * 32;                                              \
    *(uint4*)(sA[P] + lA0) = *(const uint4*)(Ah + aOff0 + ko);         \
    *(uint4*)(sA[P] + lA1) = *(const uint4*)(Ah + aOff1 + ko);         \
    _Pragma("unroll")                                                  \
    for (int c = 0; c < 4; ++c)                                        \
      *(uint4*)(sB[P] + lB[c]) = *(const uint4*)(Bp + bOff[c] + ko);   \
  }

#define QL_COMPUTE(P)                                                  \
  {                                                                    \
    const _Float16* dA = sA[P];                                        \
    const _Float16* dB = sB[P];                                        \
    Frag bfr[4];                                                       \
    _Pragma("unroll")                                                  \
    for (int nt = 0; nt < 4; ++nt) {                                   \
      const _Float16* base =                                           \
          dB + (wn * 64 + nt * 16 + lrow) * SAS + lhalf * 8;           \
      bfr[nt].q[0] = *(const uint4*)(base);                            \
      bfr[nt].q[1] = *(const uint4*)(base + 16);                       \
    }                                                                  \
    _Pragma("unroll")                                                  \
    for (int mt = 0; mt < 4; ++mt) {                                   \
      Frag af;                                                         \
      const _Float16* base =                                           \
          dA + (wm * 64 + mt * 16 + lrow) * SAS + lhalf * 8;           \
      af.q[0] = *(const uint4*)(base);                                 \
      af.q[1] = *(const uint4*)(base + 16);                            \
      _Pragma("unroll")                                                \
      for (int nt = 0; nt < 4; ++nt)                                   \
        acc[mt][nt] = __builtin_amdgcn_wmma_f32_16x16x32_f16(          \
            false, af.h, false, bfr[nt].h, (short)0, acc[mt][nt],      \
            false, false);                                             \
    }                                                                  \
  }

#if QL_ASYNC
  QL_FETCH(0, 0)                       // prologue: K-step 0 -> buffer 0
#pragma unroll 1
  for (int kb2 = 0; kb2 < KSTEPS; kb2 += 2) {
    // Phase 0: consume buffer 0, prefetch step kb2+1 into buffer 1.
    ql_wait_async0();
    __syncthreads();
    QL_FETCH(kb2 + 1, 1)               // kb2+1 <= 127 always valid
    QL_COMPUTE(0)
    // Phase 1: consume buffer 1, prefetch step kb2+2 into buffer 0.
    ql_wait_async0();
    __syncthreads();
    if (kb2 + 2 < KSTEPS) QL_FETCH(kb2 + 2, 0)
    QL_COMPUTE(1)
  }
#else
#pragma unroll 1
  for (int kb2 = 0; kb2 < KSTEPS; kb2 += 2) {
    QL_FETCH_SYNC(kb2, 0)
    __syncthreads();
    QL_COMPUTE(0)
    __syncthreads();
    QL_FETCH_SYNC(kb2 + 1, 1)
    __syncthreads();
    QL_COMPUTE(1)
    if (kb2 + 2 < KSTEPS) __syncthreads();
  }
#endif

  // ---- Fused LSTM epilogue ----------------------------------------------
  // C tile layout: lane holds N=lane%16 (= gate*4+comp), rows M = v + 8*(lane/16).
  const int gl  = lrow >> 2;                  // this lane's gate
  const int ccb = lane & 3;                   // this lane's component
  const float* bptr = (gl == 0) ? bi : (gl == 1) ? bf : (gl == 2) ? bo : bc;
  const int half = lane & 16;

#pragma unroll
  for (int nt = 0; nt < 4; ++nt) {
    int o = nb * 16 + wn * 4 + nt;
    float bias = bptr[(o << 2) + ccb];        // U-path bias for (gate, comp, o)
#pragma unroll
    for (int mt = 0; mt < 4; ++mt) {
#pragma unroll
      for (int v = 0; v < 8; ++v) {
        float pre = acc[mt][nt][v] + bias;
        // gather the 4 gates for this output across lanes (same row half)
        float pi = __shfl(pre, half + 0);
        float pf = __shfl(pre, half + 4);
        float po = __shfl(pre, half + 8);
        float pc = __shfl(pre, half + 12 + ccb);
        if (lrow < 4) {                       // lanes 0..3 / 16..19 emit cc=0..3
          int b = mb * 128 + wm * 64 + mt * 16 + v + (half ? 8 : 0);
          float ig = sigm(pi), fg = sigm(pf), og = sigm(po);
          float ct = tanhf(pc);
          int idx = ((b << 9) + o) * 4 + ccb;
          float cn = fg * c_in[idx] + ig * ct;
          out_c[idx] = cn;
          out_h[idx] = og * tanhf(cn);
        }
      }
    }
  }
}

// ---------------------------------------------------------------------------
extern "C" void kernel_launch(void* const* d_in, const int* in_sizes, int n_in,
                              void* d_out, int out_size, void* d_ws, size_t ws_size,
                              hipStream_t stream) {
  const float* x  = (const float*)d_in[0];
  const float* h  = (const float*)d_in[1];
  const float* c  = (const float*)d_in[2];
  const float* Wi = (const float*)d_in[3];
  const float* Wf = (const float*)d_in[4];
  const float* Wo = (const float*)d_in[5];
  const float* Wc = (const float*)d_in[6];
  const float* Ui = (const float*)d_in[7];
  const float* Uf = (const float*)d_in[8];
  const float* Uo = (const float*)d_in[9];
  const float* Uc = (const float*)d_in[10];
  const float* bi = (const float*)d_in[11];
  const float* bf = (const float*)d_in[12];
  const float* bo = (const float*)d_in[13];
  const float* bc = (const float*)d_in[14];

  _Float16* Ah = (_Float16*)d_ws;                                   // 16 MB
  _Float16* Bp = (_Float16*)((char*)d_ws + (size_t)BQ * KTOT * 2);  // 64 MB

  float* out_h = (float*)d_out;
  float* out_c = out_h + (size_t)BQ * HQ * 4;

  qlstm_pack_a<<<(BQ * KTOT) / 256, 256, 0, stream>>>(x, h, Ah);
  qlstm_pack_b<<<(NTOT * (size_t)KTOT) / 256, 256, 0, stream>>>(
      Wi, Wf, Wo, Wc, Ui, Uf, Uo, Uc, Bp);

  dim3 grid(NTOT / 256, BQ / 128);   // 32 x 16
  qlstm_gemm<<<grid, 256, 0, stream>>>(Ah, Bp, c, bi, bf, bo, bc, out_h, out_c);
}
